// LatticeLayer_28836410425743
// MI455X (gfx1250) — matrix operand order repaired and checked by reading, not compiled, not run on hardware
//
#include <hip/hip_runtime.h>
#include <hip/hip_bf16.h>
#include <stdint.h>

typedef __attribute__((ext_vector_type(16))) __bf16 v16bf;
typedef __attribute__((ext_vector_type(8)))  __bf16 v8bf;
typedef __attribute__((ext_vector_type(8)))  float  v8f;
typedef __attribute__((ext_vector_type(4)))  unsigned int u32x4;
typedef __attribute__((ext_vector_type(8)))  int i32x8;
typedef __attribute__((ext_vector_type(4)))  int i32x4;

#define NEGV (-10000000000.0f)

// problem constants
constexpr int B_ = 8, L_ = 512, W_ = 8, H_ = 512, E_ = 512;
constexpr int NTOK = B_ * L_;   // 4096
constexpr int QKVF = 3 * H_;    // 1536
constexpr int KDIM = 512;       // every GEMM here has K = 512

// workspace byte offsets (total ~72 MB)
constexpr size_t WS_HID_BF  = 0;                                      // 4096x512 bf16
constexpr size_t WS_HIDW_BF = WS_HID_BF  + (size_t)NTOK * H_ * 2;     // 1536x512 bf16
constexpr size_t WS_WKT_BF  = WS_HIDW_BF + (size_t)QKVF * H_ * 2;     // 3x512x512 bf16 (transposed key weights)
constexpr size_t WS_WV_BF   = WS_WKT_BF  + (size_t)3 * H_ * E_ * 2;   // 3x512x512 bf16 (value weights)
constexpr size_t WS_QKV_F32 = WS_WV_BF   + (size_t)3 * H_ * E_ * 2;   // 4096x1536 f32
constexpr size_t WS_Q_BF    = WS_QKV_F32 + (size_t)NTOK * QKVF * 4;   // 4096x512 bf16
constexpr size_t WS_QT_BF   = WS_Q_BF    + (size_t)NTOK * H_ * 2;     // 3x4096x512 bf16
constexpr size_t WS_Z_BF    = WS_QT_BF   + (size_t)3 * NTOK * E_ * 2; // 3x4096x512 bf16
constexpr size_t WS_OB_F32  = WS_Z_BF    + (size_t)3 * NTOK * E_ * 2; // 4096x512 f32

// ---------------------------------------------------------------- helpers

__device__ __forceinline__ float wave_red(float v) {
#pragma unroll
  for (int o = 16; o > 0; o >>= 1) v += __shfl_down(v, o, 32);
  return v;
}

// --- TDM staging of a B panel (rows x rowlen bf16, row stride ld elems) into LDS.
// D# layout per CDNA5 ISA 8.3/8.4: group0 = {count|..., lds_addr, global_addr, type=2},
// group1 = {mask/data_size, tensor dims, tile dims, dim0 stride}. Groups 2/3 zero (2-D).
#if __has_builtin(__builtin_amdgcn_tensor_load_to_lds)
__device__ __forceinline__ void tdm_load_panel(const __bf16* gsrc, unsigned lds_byte_addr,
                                               unsigned rows, unsigned rowlen, unsigned ld) {
  unsigned long long ga = (unsigned long long)(uintptr_t)gsrc;
  u32x4 g0;
  g0[0] = 1u;                                   // count=1, user descriptor, no gather
  g0[1] = lds_byte_addr;                        // LDS destination (byte address)
  g0[2] = (unsigned)(ga & 0xFFFFFFFFu);         // global_addr[31:0]
  g0[3] = (unsigned)((ga >> 32) & 0x01FFFFFFu) | (2u << 30);  // addr[56:32] | type=2
  i32x8 g1;
  g1[0] = (int)(1u << 16);                      // workgroup_mask=0, data_size=1 (2-byte)
  g1[1] = (int)((rowlen & 0xFFFFu) << 16);      // tensor_dim0[15:0] in bits 63:48
  g1[2] = (int)((rowlen >> 16) | ((rows & 0xFFFFu) << 16));   // dim0 hi | tensor_dim1 lo
  g1[3] = (int)((rows >> 16) | ((rowlen & 0xFFFFu) << 16));   // dim1 hi | tile_dim0
  g1[4] = (int)(rows & 0xFFFFu);                // tile_dim1 (tile_dim2 = 0)
  g1[5] = (int)ld;                              // tensor_dim0_stride[31:0]
  g1[6] = 0;
  g1[7] = 0;
  i32x4 z4 = {0, 0, 0, 0};
#if __clang_major__ >= 23
  i32x8 z8 = {0, 0, 0, 0, 0, 0, 0, 0};
  __builtin_amdgcn_tensor_load_to_lds(g0, g1, z4, z4, z8, 0);
#else
  __builtin_amdgcn_tensor_load_to_lds(g0, g1, z4, z4, 0);
#endif
}
#endif

// Stage rows x rowlen bf16 panel (row stride ld) into sB. Ends with a block barrier.
__device__ __forceinline__ void stage_panel(const __bf16* gsrc, __bf16* sB,
                                            int rows, int rowlen, int ld) {
#if __has_builtin(__builtin_amdgcn_tensor_load_to_lds)
  if (threadIdx.x < 32) {                       // one wave issues the DMA (EXEC ignored by TDM)
    tdm_load_panel(gsrc, (unsigned)(uintptr_t)sB, (unsigned)rows, (unsigned)rowlen, (unsigned)ld);
    __builtin_amdgcn_s_wait_tensorcnt(0);
  }
  __syncthreads();
#else
  const int rv = rowlen / 8;
  for (int i = threadIdx.x; i < rows * rv; i += blockDim.x) {
    const int r = i / rv, c = i - r * rv;
    *((v8bf*)sB + i) = *(const v8bf*)(gsrc + (size_t)r * ld + c * 8);
  }
  __syncthreads();
#endif
}

// Fragment loaders (bf16 16x16x32), this code's k order, same for A and B:
//   A: lane l holds row l%16, contiguous 16-K run selected by lane half
//   B: lane l holds column l%16, contiguous 16-K run selected by lane half
__device__ __forceinline__ v16bf load_a_frag(const __bf16* ar, int kk, int half) {
  union { v16bf v; v8bf h[2]; } fa;
  fa.h[0] = *(const v8bf*)(ar + kk + half * 16);
  fa.h[1] = *(const v8bf*)(ar + kk + half * 16 + 8);
  return fa.v;
}
__device__ __forceinline__ v16bf load_b_frag(const __bf16* br, int kk) {
  union { v16bf v; v8bf h[2]; } fb;
  fb.h[0] = *(const v8bf*)(br + kk);
  fb.h[1] = *(const v8bf*)(br + kk + 8);
  return fb.v;
}

// One wave computes a 16x64 strip: A row fragment (global) reused across 4 B tiles (LDS).
__device__ __forceinline__ void wave_gemm_16x64(v8f acc[4], const __bf16* A, int lda,
                                                const __bf16* sB, int m0, int lane) {
  const int half = lane >> 4;
  const __bf16* ar = A + (size_t)(m0 + (lane & 15)) * lda;
  const __bf16* br0 = sB + (0 * 16 + (lane & 15)) * KDIM + half * 16;
  const __bf16* br1 = sB + (1 * 16 + (lane & 15)) * KDIM + half * 16;
  const __bf16* br2 = sB + (2 * 16 + (lane & 15)) * KDIM + half * 16;
  const __bf16* br3 = sB + (3 * 16 + (lane & 15)) * KDIM + half * 16;
  for (int kk = 0; kk < KDIM; kk += 32) {
    v16bf fa = load_a_frag(ar, kk, half);
    acc[0] = __builtin_amdgcn_wmma_f32_16x16x32_bf16(false, fa, false, load_b_frag(br0, kk),
                                                     (short)0, acc[0], false, false);
    acc[1] = __builtin_amdgcn_wmma_f32_16x16x32_bf16(false, fa, false, load_b_frag(br1, kk),
                                                     (short)0, acc[1], false, false);
    acc[2] = __builtin_amdgcn_wmma_f32_16x16x32_bf16(false, fa, false, load_b_frag(br2, kk),
                                                     (short)0, acc[2], false, false);
    acc[3] = __builtin_amdgcn_wmma_f32_16x16x32_bf16(false, fa, false, load_b_frag(br3, kk),
                                                     (short)0, acc[3], false, false);
  }
}

// ---------------------------------------------------------------- prep kernels

__global__ void k_cvt_bf16(const float* __restrict__ src, __bf16* __restrict__ dst, int n) {
  int i = blockIdx.x * blockDim.x + threadIdx.x;
  if (i < n) dst[i] = (__bf16)src[i];
}

// dst[e][f] = src[f][e]  (512x512), f32 -> bf16
__global__ void k_transpose_cvt(const float* __restrict__ src, __bf16* __restrict__ dst) {
  int i = blockIdx.x * blockDim.x + threadIdx.x;
  int e = i >> 9, f = i & 511;
  dst[(size_t)e * 512 + f] = (__bf16)src[(size_t)f * 512 + e];
}

// ---------------------------------------------------------------- GEMM kernels
// Block tile 128(M) x 64(N): 8 waves, each 16x64; shared 64x512 bf16 B panel in LDS
// staged by the Tensor Data Mover.

// QKV = hiddens(4096x512) . hid_W(1536x512)^T + hid_b ; f32 out, first 512 cols also bf16
__global__ __launch_bounds__(256) void k_gemm_qkv(const __bf16* __restrict__ A,
                                                  const __bf16* __restrict__ Bm,
                                                  const float* __restrict__ bias,
                                                  float* __restrict__ C32,
                                                  __bf16* __restrict__ Qbf) {
  __shared__ __bf16 sB[64 * KDIM];
  const int blocks_n = QKVF / 64;  // 24
  const int bm = blockIdx.x / blocks_n, bn = blockIdx.x % blocks_n;
  const int lane = threadIdx.x & 31;
  const int m0 = bm * 128 + (threadIdx.x >> 5) * 16;
  const int nb = bn * 64;

  stage_panel(Bm + (size_t)nb * KDIM, sB, 64, KDIM, KDIM);

  v8f acc[4] = {};
  wave_gemm_16x64(acc, A, H_, sB, m0, lane);

#pragma unroll
  for (int tn = 0; tn < 4; ++tn) {
    const int col = nb + tn * 16 + (lane & 15);
    const float b = bias[col];
#pragma unroll
    for (int r = 0; r < 8; ++r) {
      const int row = m0 + r + (lane >> 4) * 8;
      const float v = acc[tn][r] + b;
      C32[(size_t)row * QKVF + col] = v;
      if (col < H_) Qbf[(size_t)row * H_ + col] = (__bf16)v;
    }
  }
}

// QT_t = Q(4096x512) . WkT_t(512x512)^T   (t = blockIdx.y)
__global__ __launch_bounds__(256) void k_gemm_qt(const __bf16* __restrict__ Q,
                                                 const __bf16* __restrict__ WkT,
                                                 __bf16* __restrict__ QT) {
  __shared__ __bf16 sB[64 * KDIM];
  const int t = blockIdx.y;
  const __bf16* Bm = WkT + (size_t)t * H_ * E_;
  __bf16* out = QT + (size_t)t * NTOK * E_;
  const int blocks_n = E_ / 64;  // 8
  const int bm = blockIdx.x / blocks_n, bn = blockIdx.x % blocks_n;
  const int lane = threadIdx.x & 31;
  const int m0 = bm * 128 + (threadIdx.x >> 5) * 16;
  const int nb = bn * 64;

  stage_panel(Bm + (size_t)nb * KDIM, sB, 64, KDIM, KDIM);

  v8f acc[4] = {};
  wave_gemm_16x64(acc, Q, H_, sB, m0, lane);

#pragma unroll
  for (int tn = 0; tn < 4; ++tn) {
    const int col = nb + tn * 16 + (lane & 15);
#pragma unroll
    for (int r = 0; r < 8; ++r) {
      const int row = m0 + r + (lane >> 4) * 8;
      out[(size_t)row * E_ + col] = (__bf16)acc[tn][r];
    }
  }
}

// out = base + sum_t Z_t(4096x512) . Wv_t(512x512)^T  (panels restaged per t)
__global__ __launch_bounds__(256) void k_gemm_out(const __bf16* __restrict__ Z,
                                                  const __bf16* __restrict__ Wv,
                                                  const float* __restrict__ base,
                                                  float* __restrict__ out) {
  __shared__ __bf16 sB[64 * KDIM];
  const int blocks_n = H_ / 64;  // 8
  const int bm = blockIdx.x / blocks_n, bn = blockIdx.x % blocks_n;
  const int lane = threadIdx.x & 31;
  const int m0 = bm * 128 + (threadIdx.x >> 5) * 16;
  const int nb = bn * 64;

  v8f acc[4] = {};
  for (int t = 0; t < 3; ++t) {
    __syncthreads();  // previous panel fully consumed
    stage_panel(Wv + (size_t)t * H_ * E_ + (size_t)nb * KDIM, sB, 64, KDIM, KDIM);
    wave_gemm_16x64(acc, Z + (size_t)t * NTOK * E_, E_, sB, m0, lane);
  }

#pragma unroll
  for (int tn = 0; tn < 4; ++tn) {
    const int col = nb + tn * 16 + (lane & 15);
#pragma unroll
    for (int r = 0; r < 8; ++r) {
      const int row = m0 + r + (lane >> 4) * 8;
      out[(size_t)row * H_ + col] = acc[tn][r] + base[(size_t)row * H_ + col];
    }
  }
}

// ---------------------------------------------------------------- attention kernel
// One workgroup per token. Embeddings gathered once into LDS (24x512 f32 = 48KB),
// reused for both score dots and z accumulation.
__global__ __launch_bounds__(256) void k_attn(
    const float* __restrict__ QKV, const __bf16* __restrict__ QT,
    const unsigned char* __restrict__ hmask,
    const int* __restrict__ begins, const unsigned char* __restrict__ bmask,
    const int* __restrict__ middles, const unsigned char* __restrict__ mmask,
    const int* __restrict__ ends, const unsigned char* __restrict__ emask,
    const float* __restrict__ W_emb,
    const float* __restrict__ beg_b, const float* __restrict__ mid_b,
    const float* __restrict__ end_b,
    __bf16* __restrict__ Z, float* __restrict__ OB) {
  __shared__ float s_emb[24][512];
  __shared__ float s_score[32];
  __shared__ float s_red[4];
  __shared__ float s_w[32];
  __shared__ float s_wsum[4];

  const int n = blockIdx.x;
  const int tid = threadIdx.x;
  const int lane = tid & 31;

  if (tid < 32) s_score[tid] = 0.f;
  if (tid < 4) { s_red[tid] = 0.f; s_wsum[tid] = 0.f; }
  __syncthreads();

  // phase 1: s0 = q.hk ; qb_t = q.b_k_t (key-bias folded through the rewrite)
  const float* qkvn = QKV + (size_t)n * QKVF;
  float a0 = 0.f, a1 = 0.f, a2 = 0.f, a3 = 0.f;
#pragma unroll
  for (int rep = 0; rep < 2; ++rep) {
    const int i = tid + rep * 256;
    const float q = qkvn[i];
    a0 += q * qkvn[H_ + i];
    a1 += q * beg_b[i];
    a2 += q * mid_b[i];
    a3 += q * end_b[i];
  }
  a0 = wave_red(a0); a1 = wave_red(a1); a2 = wave_red(a2); a3 = wave_red(a3);
  if (lane == 0) {
    atomicAdd(&s_red[0], a0); atomicAdd(&s_red[1], a1);
    atomicAdd(&s_red[2], a2); atomicAdd(&s_red[3], a3);
  }

  // phase 2: gather embeddings into LDS + score dots  s_j = qt_t . e_j
  const int* idxp[3] = {begins, middles, ends};
  for (int t = 0; t < 3; ++t) {
    const __bf16* qtn = QT + ((size_t)t * NTOK + n) * E_;
    const float qt0 = (float)qtn[tid];
    const float qt1 = (float)qtn[tid + 256];
    for (int w = 0; w < 8; ++w) {
      const int j = t * 8 + w;
      const int idx = idxp[t][(size_t)n * W_ + w];
      const float* ep = W_emb + (size_t)idx * E_;
      const float e0 = ep[tid], e1 = ep[tid + 256];
      s_emb[j][tid] = e0;
      s_emb[j][tid + 256] = e1;
      const float p = wave_red(qt0 * e0 + qt1 * e1);
      if (lane == 0) atomicAdd(&s_score[j], p);
    }
  }
  __syncthreads();

  // phase 3: masked softmax over 25 scores (mask true -> NEG, same as reference)
  if (tid == 0) {
    const unsigned char* mkp[3] = {bmask, mmask, emask};
    float sc[25];
    sc[0] = hmask[n] ? NEGV : s_red[0];
    for (int j = 0; j < 24; ++j) {
      const int t = j >> 3;
      sc[1 + j] = mkp[t][(size_t)n * W_ + (j & 7)] ? NEGV : (s_score[j] + s_red[1 + t]);
    }
    float mx = sc[0];
    for (int k = 1; k < 25; ++k) mx = fmaxf(mx, sc[k]);
    float sum = 0.f;
    for (int k = 0; k < 25; ++k) { sc[k] = __expf(sc[k] - mx); sum += sc[k]; }
    const float inv = 1.f / sum;
    for (int k = 0; k < 25; ++k) s_w[k] = sc[k] * inv;
    for (int t = 0; t < 3; ++t) {
      float wsm = 0.f;
      for (int w = 0; w < 8; ++w) wsm += s_w[1 + t * 8 + w];
      s_wsum[t] = wsm;
    }
  }
  __syncthreads();

  // phase 4: z_t = sum_j w_j e_j (bf16 for the output GEMM) and value-bias base
  const float w0 = s_w[0];
  const float ws0 = s_wsum[0], ws1 = s_wsum[1], ws2 = s_wsum[2];
#pragma unroll
  for (int rep = 0; rep < 2; ++rep) {
    const int e = tid + rep * 256;
    for (int t = 0; t < 3; ++t) {
      float z = 0.f;
#pragma unroll
      for (int w = 0; w < 8; ++w) z += s_w[1 + t * 8 + w] * s_emb[t * 8 + w][e];
      Z[((size_t)t * NTOK + n) * E_ + e] = (__bf16)z;
    }
    float obv = w0 * qkvn[2 * H_ + e];
    obv += ws0 * beg_b[H_ + e] + ws1 * mid_b[H_ + e] + ws2 * end_b[H_ + e];
    OB[(size_t)n * H_ + e] = obv;
  }
}

// ---------------------------------------------------------------- launch

extern "C" void kernel_launch(void* const* d_in, const int* in_sizes, int n_in,
                              void* d_out, int out_size, void* d_ws, size_t ws_size,
                              hipStream_t stream) {
  const float* hiddens = (const float*)d_in[0];
  const unsigned char* masks = (const unsigned char*)d_in[1];
  const int* begins = (const int*)d_in[2];
  const unsigned char* begin_masks = (const unsigned char*)d_in[3];
  const int* middles = (const int*)d_in[4];
  const unsigned char* middle_masks = (const unsigned char*)d_in[5];
  const int* ends = (const int*)d_in[6];
  const unsigned char* end_masks = (const unsigned char*)d_in[7];
  const float* W_emb = (const float*)d_in[8];
  const float* hid_W = (const float*)d_in[9];
  const float* hid_b = (const float*)d_in[10];
  const float* beg_W = (const float*)d_in[11];
  const float* beg_b = (const float*)d_in[12];
  const float* mid_W = (const float*)d_in[13];
  const float* mid_b = (const float*)d_in[14];
  const float* end_W = (const float*)d_in[15];
  const float* end_b = (const float*)d_in[16];

  char* ws = (char*)d_ws;
  __bf16* hid_bf  = (__bf16*)(ws + WS_HID_BF);
  __bf16* hidw_bf = (__bf16*)(ws + WS_HIDW_BF);
  __bf16* wkt_bf  = (__bf16*)(ws + WS_WKT_BF);
  __bf16* wv_bf   = (__bf16*)(ws + WS_WV_BF);
  float*  qkv     = (float*)(ws + WS_QKV_F32);
  __bf16* q_bf    = (__bf16*)(ws + WS_Q_BF);
  __bf16* qt_bf   = (__bf16*)(ws + WS_QT_BF);
  __bf16* z_bf    = (__bf16*)(ws + WS_Z_BF);
  float*  ob      = (float*)(ws + WS_OB_F32);

  // 1) bf16 conversion of activations/weights; transpose key weights (all GEMMs become NT)
  k_cvt_bf16<<<(NTOK * H_) / 256, 256, 0, stream>>>(hiddens, hid_bf, NTOK * H_);
  k_cvt_bf16<<<(QKVF * H_) / 256, 256, 0, stream>>>(hid_W, hidw_bf, QKVF * H_);
  const float* wsrc[3] = {beg_W, mid_W, end_W};
  for (int t = 0; t < 3; ++t) {
    k_transpose_cvt<<<(H_ * E_) / 256, 256, 0, stream>>>(wsrc[t], wkt_bf + (size_t)t * H_ * E_);
    k_cvt_bf16<<<(H_ * E_) / 256, 256, 0, stream>>>(wsrc[t] + (size_t)H_ * E_,
                                                    wv_bf + (size_t)t * H_ * E_, H_ * E_);
  }

  // 2) QKV projection GEMM: 128x64 block tiles, TDM-staged B panels
  k_gemm_qkv<<<(NTOK / 128) * (QKVF / 64), 256, 0, stream>>>(hid_bf, hidw_bf, hid_b, qkv, q_bf);

  // 3) QT_t = Q . Wk_t  (three GEMMs via blockIdx.y)
  dim3 gq((NTOK / 128) * (E_ / 64), 3);
  k_gemm_qt<<<gq, 256, 0, stream>>>(q_bf, wkt_bf, qt_bf);

  // 4) per-token gather + scores + softmax + z accumulation
  k_attn<<<NTOK, 256, 0, stream>>>(qkv, qt_bf, masks, begins, begin_masks, middles,
                                   middle_masks, ends, end_masks, W_emb,
                                   beg_b, mid_b, end_b, z_bf, ob);

  // 5) out = base + sum_t Z_t . Wv_t^T
  k_gemm_out<<<(NTOK / 128) * (H_ / 64), 256, 0, stream>>>(z_bf, wv_bf, ob, (float*)d_out);
}
